// ConditionalAutoregressiveFlow_64699387346973
// MI455X (gfx1250) — matrix operand order repaired and checked by reading, not compiled, not run on hardware
//
#include <hip/hip_runtime.h>
#include <hip/hip_fp16.h>

typedef __attribute__((ext_vector_type(16))) _Float16 v16h;
typedef __attribute__((ext_vector_type(8)))  float    v8f;

#define T_STEPS 2048
#define HID     512
#define G4      (4 * HID)
#define WPG     16        // workgroups per batch-group (hidden slices)
#define NGRP    8         // batch groups of 16 rows
#define UPW     32        // hidden units per WG
#define THREADS 256       // 8 waves

__device__ __forceinline__ float fast_sigmoid(float z) {
    return __builtin_amdgcn_rcpf(1.0f + __expf(-z));
}
__device__ __forceinline__ float fast_tanh(float z) {
    return 1.0f - 2.0f * __builtin_amdgcn_rcpf(__expf(2.0f * z) + 1.0f);
}

__launch_bounds__(THREADS)
__global__ void rev_lstm_wmma_kernel(const float* __restrict__ x,
                                     const float* __restrict__ Wi,
                                     const float* __restrict__ Wh,
                                     const float* __restrict__ bias,
                                     float* __restrict__ out,
                                     _Float16* __restrict__ hbuf,   // [2][NGRP][16*HID] f16
                                     unsigned* __restrict__ cnt)    // [NGRP] monotonic
{
    __shared__ __align__(16) _Float16 sH[16 * HID];   // staged h tile (16 KB)
    __shared__ float sZ[8 * 16 * 16];                 // z tiles, one per wave (8 KB)
    __shared__ float sX[16];                          // x values this step

    const int wg   = blockIdx.x;
    const int g    = wg / WPG;          // batch group (rows 16g..16g+15)
    const int w    = wg % WPG;          // hidden slice (units 32w..32w+31)
    const int tid  = threadIdx.x;
    const int lane = tid & 31;
    const int wave = tid >> 5;          // == n-tile id 0..7 (gate q = wave/2)

    // ---------------- one-time: Wh slice -> f16 B fragments in VGPRs ----------------
    const int qg      = wave >> 1;
    const int colbase = qg * HID + w * UPW + (wave & 1) * 16;
    const int ncol    = colbase + (lane & 15);
    const int koffB   = (lane < 16) ? 0 : 8;
    v16h bfrag[16];
#pragma unroll
    for (int kk = 0; kk < 16; ++kk) {
        v16h tmp;
#pragma unroll
        for (int i = 0; i < 16; ++i)
            tmp[i] = (_Float16)Wh[(size_t)(kk * 32 + koffB + (i < 8 ? i : i + 8)) * G4 + ncol];
        bfrag[kk] = tmp;
    }

    // ---------------- one-time: per-thread elementwise constants ----------------
    // 16 rows x 32 units = 512 elements; 2 per thread (same row m, units j, j+1).
    float c_st[2];
    float wi_r[2][4], b_r[2][4];
    const int m_el = (tid * 2) >> 5;        // row for both elements
    const int j0   = (tid * 2) & 31;        // even unit; j1 = j0+1
#pragma unroll
    for (int e = 0; e < 2; ++e) {
        c_st[e] = 0.0f;
#pragma unroll
        for (int q = 0; q < 4; ++q) {
            const int c4 = q * HID + w * UPW + (j0 + e);
            wi_r[e][q] = Wi[c4];
            b_r[e][q]  = bias[c4];
        }
    }

    _Float16* hslot0 = hbuf + (size_t)(0 * NGRP + g) * (16 * HID);
    _Float16* hslot1 = hbuf + (size_t)(1 * NGRP + g) * (16 * HID);
    const float* xrow = x + (size_t)(g * 16) * T_STEPS;

    // A-fragment addressing (16x32 f16 A): lane holds row m=lane%16;
    // lanes<16 carry K {0..7,16..23}, lanes>=16 carry K {8..15,24..31}.
    const _Float16* arow = sH + (size_t)(lane & 15) * HID;
    const int o8 = (lane < 16) ? 0 : 8;

    auto loadA = [&](int kk) {
        v16h a;
        *((uint4*)&a)       = *(const uint4*)(arow + kk * 32 + o8);
        *(((uint4*)&a) + 1) = *(const uint4*)(arow + kk * 32 + 16 + o8);
        return a;
    };

    // LDS byte offset of sH (generic LDS pointer: low 32 bits = LDS offset)
    const unsigned sH_lds = (unsigned)(uintptr_t)(void*)sH;

    for (int t = 0; t < T_STEPS; ++t) {
        const int p = t & 1;
        _Float16* hin  = p ? hslot1 : hslot0;
        _Float16* hout = p ? hslot0 : hslot1;

        // ---- stage h_t (16x512 f16 = 16KB) via async global->LDS DMA ----
        // SCOPE_DEV: h was produced by other WGs; must not hit stale WGP$ lines.
        {
#pragma unroll
            for (int i = 0; i < 4; ++i) {
                const unsigned c     = (unsigned)tid + i * THREADS;      // uint4 index
                const unsigned laddr = sH_lds + c * 16u;
                const unsigned long long gaddr =
                    (unsigned long long)(const void*)((const uint4*)hin + c);
                asm volatile("global_load_async_to_lds_b128 %0, %1, off scope:SCOPE_DEV"
                             :: "v"(laddr), "v"(gaddr) : "memory");
            }
            if (tid < 16) sX[tid] = xrow[(size_t)tid * T_STEPS + (T_STEPS - 1 - t)];
            if (tid < 16 && t + 8 < T_STEPS)
                __builtin_prefetch(&xrow[(size_t)tid * T_STEPS + (T_STEPS - 9 - t)], 0, 1);
            asm volatile("s_wait_asynccnt 0x0" ::: "memory");
        }
        __syncthreads();

        // ---- z tile = h_t (16x512) * Wh_tile (512x16), f16 WMMA, f32 accum ----
        // Depth-4 A-fragment ring, pinned with sched_barrier so the scheduler
        // cannot re-serialize ds_load -> wmma (keeps ~8 ds loads in flight).
        v8f acc = {0.f, 0.f, 0.f, 0.f, 0.f, 0.f, 0.f, 0.f};
        {
            v16h af[4];
#pragma unroll
            for (int i = 0; i < 4; ++i) af[i] = loadA(i);
            __builtin_amdgcn_sched_barrier(0);
#pragma unroll
            for (int kk = 0; kk < 16; ++kk) {
                acc = __builtin_amdgcn_wmma_f32_16x16x32_f16(
                          false, af[kk & 3], false, bfrag[kk], (short)0, acc,
                          false, false);
                if (kk + 4 < 16) af[kk & 3] = loadA(kk + 4);
                __builtin_amdgcn_sched_barrier(0);
            }
        }

        // ---- spill z tile to LDS for cross-wave gate combine ----
        // C layout: VGPR r, lane l -> m = r + 8*(l/16), n = l%16
        {
            float* zt = sZ + wave * 256;
            const int nloc = lane & 15;
            const int mhi  = (lane >> 4) * 8;
#pragma unroll
            for (int r = 0; r < 8; ++r)
                zt[(mhi + r) * 16 + nloc] = acc[r];
        }
        __syncthreads();

        // ---- gates + state update (c in registers), packed stores ----
        {
            const float xv = sX[m_el];
            float hv[2];
#pragma unroll
            for (int e = 0; e < 2; ++e) {
                const int j = j0 + e;
                float z4[4];
#pragma unroll
                for (int q = 0; q < 4; ++q) {
                    const int tt = q * 2 + (j >> 4);
                    z4[q] = sZ[tt * 256 + m_el * 16 + (j & 15)]
                            + xv * wi_r[e][q] + b_r[e][q];
                }
                const float ig = fast_sigmoid(z4[0]);
                const float fg = fast_sigmoid(z4[1]);
                const float gg = fast_tanh(z4[2]);
                const float og = fast_sigmoid(z4[3]);
                const float cv = fg * c_st[e] + ig * gg;
                c_st[e] = cv;
                hv[e] = og * fast_tanh(cv);
            }
            // one b32 store for the f16 pair, one b64 store for the f32 pair
            const int unit0 = w * UPW + j0;                   // even
            const unsigned short u0 = __builtin_bit_cast(unsigned short, (_Float16)hv[0]);
            const unsigned short u1 = __builtin_bit_cast(unsigned short, (_Float16)hv[1]);
            *(unsigned*)&hout[m_el * HID + unit0] = (unsigned)u0 | ((unsigned)u1 << 16);
            float2 o2; o2.x = hv[0]; o2.y = hv[1];
            *(float2*)&out[((size_t)(g * 16 + m_el) * T_STEPS + t) * HID + unit0] = o2;
        }

        // ---- make h_{t+1} visible, then group-wide barrier via L2 atomics ----
        __threadfence();
        __syncthreads();
        if (t + 1 < T_STEPS) {
            if (tid == 0) {
                __hip_atomic_fetch_add(&cnt[g], 1u, __ATOMIC_RELEASE,
                                       __HIP_MEMORY_SCOPE_AGENT);
                const unsigned target = (unsigned)(WPG * (t + 1));
                while (__hip_atomic_load(&cnt[g], __ATOMIC_ACQUIRE,
                                         __HIP_MEMORY_SCOPE_AGENT) < target)
                    __builtin_amdgcn_s_sleep(1);
            }
            __syncthreads();
        }
    }
}

extern "C" void kernel_launch(void* const* d_in, const int* in_sizes, int n_in,
                              void* d_out, int out_size, void* d_ws, size_t ws_size,
                              hipStream_t stream) {
    // inputs: 0=s (unused), 1=x (128x2048 f32), 2=Wi (1x2048), 3=Wh (512x2048), 4=b (2048)
    const float* x  = (const float*)d_in[1];
    const float* Wi = (const float*)d_in[2];
    const float* Wh = (const float*)d_in[3];
    const float* b  = (const float*)d_in[4];
    float* out = (float*)d_out;

    // workspace: [0, 256KB) = double-buffered h (f16), then 8 barrier counters
    const size_t hbuf_bytes = (size_t)2 * NGRP * 16 * HID * sizeof(_Float16); // 256 KB
    _Float16* hbuf = (_Float16*)d_ws;
    unsigned* cnt  = (unsigned*)((char*)d_ws + hbuf_bytes);

    // h0 = 0, counters = 0 (graph-capturable async memset on the stream)
    hipMemsetAsync(d_ws, 0, hbuf_bytes + NGRP * sizeof(unsigned), stream);

    dim3 grid(NGRP * WPG);   // 128 persistent workgroups
    dim3 block(THREADS);     // 8 waves of 32
    rev_lstm_wmma_kernel<<<grid, block, 0, stream>>>(x, Wi, Wh, b, out, hbuf, cnt);
}
